// MaxHybridFlatten_54116587929984
// MI455X (gfx1250) — matrix operand order repaired and checked by reading, not compiled, not run on hardware
//
#include <hip/hip_runtime.h>
#include <hip/hip_bf16.h>

// ---------------------------------------------------------------------------
// MaxHybridFlatten for MI455X (gfx1250, wave32, WMMA)
//
// out[b,hw,e]   = (sum_c feature[b,c,hw] * W[e,c] + bias[e]) * gate[b,hw]
// gate[b,hw]    = max_j la_outs[b,j,hw]
// keep[b,0..9]  = indices of 10 largest gate values per batch
//
// Roofline: 19.7 GFLOP GEMM vs ~206 MB traffic -> ~9us memory floor @23.3TB/s.
// Split-bf16 (hi+lo, 3 products) on v_wmma_f32_16x16x32_bf16 gives ~f32
// accuracy at ~3/8 the cost of the f32 16x16x4 WMMA path -> near memory-bound.
//
// Round-2 refinements:
//  * double-buffered LDS staging: global loads for tile k+1 issue before the
//    21 WMMAs of tile k, hiding HBM latency; one barrier per k-step.
//  * non-temporal b128 stores for the 154MB write-once output stream so the
//    L2 (192MB) keeps the feature tensor resident across the 6 e-blocks.
// ---------------------------------------------------------------------------

typedef __attribute__((ext_vector_type(16))) __bf16 v16bf;
typedef __attribute__((ext_vector_type(8)))  float  v8f;
typedef __attribute__((ext_vector_type(4)))  float  v4f;

#define WMMA_BF16(A, B, Cacc) \
  __builtin_amdgcn_wmma_f32_16x16x32_bf16(false, (A), false, (B), (short)0, (Cacc), false, false)

union Frag {
  v16bf  v;
  float4 f4[2];
};

#define BATCH 64
#define CH    256
#define HW    784   // 28*28 = 49*16
#define EMB   768   // 48*16
#define LDS_STRIDE 40  // bf16 units per hw-row (80B): 16B-aligned b128 reads,
                       // 20-bank row stride spreads LDS bank conflicts

__global__ __launch_bounds__(256)
void gemm_gate_kernel(const float* __restrict__ feature,  // (64,256,28,28)
                      const float* __restrict__ la,       // (64,4,28,28)
                      const float* __restrict__ w,        // (768,256)
                      const float* __restrict__ bias,     // (768)
                      float* __restrict__ out)            // (64,784,768)
{
  // Double-buffered staged feature tile, pre-split hi/lo bf16, [hw][c] layout.
  __shared__ __align__(16) __bf16 sHi[2][112 * LDS_STRIDE];
  __shared__ __align__(16) __bf16 sLo[2][112 * LDS_STRIDE];

  const int tid   = threadIdx.x;
  const int b     = blockIdx.z;
  const int hwBlk = blockIdx.x * 112;              // 7 blocks * 112 = 784
  const int wave  = tid >> 5;                      // 8 waves
  const int lane  = tid & 31;
  const int lhalf = lane >> 4;                     // 0: lanes 0-15, 1: 16-31
  const int ln    = lane & 15;
  const int eBase = blockIdx.y * 128 + wave * 16;  // 6 blocks * 8 * 16 = 768

  const float* featB = feature + (size_t)b * CH * HW;

  v8f acc[7];
  #pragma unroll
  for (int n = 0; n < 7; ++n) {
    v8f z = {0.f, 0.f, 0.f, 0.f, 0.f, 0.f, 0.f, 0.f};
    acc[n] = z;
  }

  // Prologue: stage tile k=0 into buffer 0.
  #pragma unroll
  for (int i = 0; i < 14; ++i) {                   // 14 * 256 = 3584 = 32*112
    const int idx = tid + i * 256;
    const int c   = idx / 112;
    const int x   = idx - c * 112;
    const float f = featB[(size_t)c * HW + hwBlk + x];
    const __bf16 h = (__bf16)f;
    sHi[0][x * LDS_STRIDE + c] = h;
    sLo[0][x * LDS_STRIDE + c] = (__bf16)(f - (float)h);
  }

  for (int k = 0; k < 8; ++k) {                    // K = 256 in steps of 32
    const int c0  = k * 32;
    const int cur = k & 1;

    __syncthreads();  // staged writes of buf[cur] visible; prior reads done

    // Issue global loads for tile k+1 now; latency hides behind the WMMAs.
    float nf[14];
    if (k < 7) {
      #pragma unroll
      for (int i = 0; i < 14; ++i) {
        const int idx = tid + i * 256;
        const int c   = idx / 112;
        const int x   = idx - c * 112;
        nf[i] = featB[(size_t)(c0 + 32 + c) * HW + hwBlk + x];
      }
    }

    // A fragment: 16x32 bf16 tile of W (e rows, c cols), split hi/lo.
    // Layout: lanes 0-15 row M=lane, K={0..7,16..23};
    //         lanes 16-31 row M=lane-16, K={8..15,24..31}.
    const float* wr = w + (size_t)(eBase + ln) * CH + c0 + lhalf * 8;
    const float4 a0 = *(const float4*)(wr + 0);
    const float4 a1 = *(const float4*)(wr + 4);
    const float4 a2 = *(const float4*)(wr + 16);
    const float4 a3 = *(const float4*)(wr + 20);
    const float av[16] = {a0.x, a0.y, a0.z, a0.w, a1.x, a1.y, a1.z, a1.w,
                          a2.x, a2.y, a2.z, a2.w, a3.x, a3.y, a3.z, a3.w};
    v16bf ah, al;
    #pragma unroll
    for (int j = 0; j < 16; ++j) {
      const __bf16 h = (__bf16)av[j];
      ah[j] = h;
      al[j] = (__bf16)(av[j] - (float)h);
    }

    #pragma unroll
    for (int n = 0; n < 7; ++n) {
      // B fragment: 32x16 bf16 (c rows, hw cols).
      // Lanes 0-15 col N=lane hold K={0..7,16..23}; lanes 16-31 hold K={8..15,24..31}.
      Frag bh, bl;
      const int off = (n * 16 + ln) * LDS_STRIDE + lhalf * 8;
      bh.f4[0] = *(const float4*)&sHi[cur][off];
      bh.f4[1] = *(const float4*)&sHi[cur][off + 16];
      bl.f4[0] = *(const float4*)&sLo[cur][off];
      bl.f4[1] = *(const float4*)&sLo[cur][off + 16];

      // split-bf16: hi*hi + hi*lo + lo*hi (lo*lo below f32 ulp)
      acc[n] = WMMA_BF16(ah, bh.v, acc[n]);
      acc[n] = WMMA_BF16(ah, bl.v, acc[n]);
      acc[n] = WMMA_BF16(al, bh.v, acc[n]);
    }

    // Convert and write tile k+1 into the other buffer (no barrier needed:
    // that buffer was last read before the barrier at the top of this iter).
    if (k < 7) {
      const int nxt = cur ^ 1;
      #pragma unroll
      for (int i = 0; i < 14; ++i) {
        const int idx = tid + i * 256;
        const int c   = idx / 112;
        const int x   = idx - c * 112;
        const __bf16 h = (__bf16)nf[i];
        sHi[nxt][x * LDS_STRIDE + c] = h;
        sLo[nxt][x * LDS_STRIDE + c] = (__bf16)(nf[i] - (float)h);
      }
    }
  }

  // Epilogue: D layout gives each lane 8 consecutive e for one hw column:
  //   vgpr r -> e = eBase + lhalf*8 + r,  hw = hw_tile + ln
  const int eOff = eBase + lhalf * 8;
  const float4 b0 = *(const float4*)(bias + eOff);
  const float4 b1 = *(const float4*)(bias + eOff + 4);

  #pragma unroll
  for (int n = 0; n < 7; ++n) {
    const int hw = hwBlk + n * 16 + ln;
    const float* lp = la + (size_t)b * 4 * HW + hw;
    const float g = fmaxf(fmaxf(lp[0], lp[HW]), fmaxf(lp[2 * HW], lp[3 * HW]));

    float* o = out + ((size_t)b * HW + hw) * EMB + eOff;
    v4f o0, o1;
    o0[0] = (acc[n][0] + b0.x) * g;  o0[1] = (acc[n][1] + b0.y) * g;
    o0[2] = (acc[n][2] + b0.z) * g;  o0[3] = (acc[n][3] + b0.w) * g;
    o1[0] = (acc[n][4] + b1.x) * g;  o1[1] = (acc[n][5] + b1.y) * g;
    o1[2] = (acc[n][6] + b1.z) * g;  o1[3] = (acc[n][7] + b1.w) * g;
    // Write-once 154MB stream: non-temporal so L2 keeps feature resident.
    __builtin_nontemporal_store(o0, (v4f*)o);
    __builtin_nontemporal_store(o1, (v4f*)o + 1);
  }
}

// Top-10 of gate per batch: iterated argmax with lowest-index tie-break
// (matches jax.lax.top_k ordering for descending values).
__global__ __launch_bounds__(256)
void topk_kernel(const float* __restrict__ la, int* __restrict__ keep)
{
  __shared__ float sg[HW];
  __shared__ float rv[256];
  __shared__ int   ri[256];

  const int b = blockIdx.x;
  const int t = threadIdx.x;

  for (int hw = t; hw < HW; hw += 256) {
    const float* lp = la + (size_t)b * 4 * HW + hw;
    sg[hw] = fmaxf(fmaxf(lp[0], lp[HW]), fmaxf(lp[2 * HW], lp[3 * HW]));
  }
  __syncthreads();

  for (int it = 0; it < 10; ++it) {
    float bv = -INFINITY;
    int   bi = 0x7fffffff;
    for (int hw = t; hw < HW; hw += 256) {
      const float v = sg[hw];
      if (v > bv || (v == bv && hw < bi)) { bv = v; bi = hw; }
    }
    rv[t] = bv; ri[t] = bi;
    __syncthreads();
    for (int s = 128; s > 0; s >>= 1) {
      if (t < s) {
        const float v2 = rv[t + s];
        const int   i2 = ri[t + s];
        if (v2 > rv[t] || (v2 == rv[t] && i2 < ri[t])) { rv[t] = v2; ri[t] = i2; }
      }
      __syncthreads();
    }
    if (t == 0) {
      keep[b * 10 + it] = ri[0];
      sg[ri[0]] = -INFINITY;   // mask out selected element
    }
    __syncthreads();
  }
}

extern "C" void kernel_launch(void* const* d_in, const int* in_sizes, int n_in,
                              void* d_out, int out_size, void* d_ws, size_t ws_size,
                              hipStream_t stream)
{
  const float* feature = (const float*)d_in[0];  // (64,256,28,28)
  const float* la      = (const float*)d_in[1];  // (64,4,28,28)
  const float* proj_w  = (const float*)d_in[2];  // (768,256)
  const float* proj_b  = (const float*)d_in[3];  // (768)

  float* outs = (float*)d_out;                               // 64*784*768 floats
  int*   keep = (int*)(outs + (size_t)BATCH * HW * EMB);     // 64*10 int32 after

  dim3 grid(7, 6, BATCH);   // 7 hw-blocks(112) x 6 e-blocks(128) x 64 batches
  gemm_gate_kernel<<<grid, 256, 0, stream>>>(feature, la, proj_w, proj_b, outs);
  topk_kernel<<<BATCH, 256, 0, stream>>>(la, keep);
}
